// Head_55516747268177
// MI455X (gfx1250) — matrix operand order repaired and checked by reading, not compiled, not run on hardware
//
#include <hip/hip_runtime.h>

#define HIDDEN 1024
#define HEAD   64
#define BATCH  4
#define TLEN   4096

typedef __bf16 bf16_t;
typedef __attribute__((ext_vector_type(16))) __bf16 v16bf;
typedef __attribute__((ext_vector_type(8)))  __bf16 v8bf;
typedef __attribute__((ext_vector_type(8)))  float  v8f;

static __device__ inline v8f wmma_bf16(v16bf a, v16bf b, v8f c) {
    // D = A(16x32 bf16) x B(32x16 bf16) + C(16x16 f32)
    return __builtin_amdgcn_wmma_f32_16x16x32_bf16(false, a, false, b,
                                                   (short)0, c, false, false);
}

static __device__ inline v16bf join16(v8bf lo, v8bf hi) {
    return __builtin_shufflevector(lo, hi, 0,1,2,3,4,5,6,7,8,9,10,11,12,13,14,15);
}

// B-matrix fragment: 16 contiguous bf16 per lane (lane = khalf*16 + n)
template <typename PTR>
static __device__ inline v16bf loadB16(PTR p) {
    v8bf lo = *(const v8bf*)(p);
    v8bf hi = *(const v8bf*)(p + 8);
    return join16(lo, hi);
}

// A-matrix fragment (16x32 bf16): lane holds K = 2*koff8..2*koff8+15 (contiguous)
template <typename PTR>
static __device__ inline v16bf loadA16(PTR p, int koff8) {
    v8bf lo = *(const v8bf*)(p + 2 * koff8);
    v8bf hi = *(const v8bf*)(p + 2 * koff8 + 8);
    return join16(lo, hi);
}

// --- CDNA5 async global->LDS (tracked by ASYNCcnt) -------------------------
static __device__ inline unsigned lds_off32(const void* p) {
    // generic ptr to LDS: addr[31:0] is the LDS byte offset (flat aperture)
    return (unsigned)(uintptr_t)p;
}
static __device__ inline void async_copy_b128(unsigned lds_off, const void* gaddr) {
    asm volatile("global_load_async_to_lds_b128 %0, %1, off"
                 :: "v"(lds_off), "v"(gaddr) : "memory");
}
static __device__ inline void wait_async_le16() {
    asm volatile("s_wait_asynccnt 0x10" ::: "memory");
}
static __device__ inline void wait_async_0() {
    asm volatile("s_wait_asynccnt 0x0" ::: "memory");
}
static __device__ inline void wait_ds0() {
    asm volatile("s_wait_dscnt 0x0" ::: "memory");
}

// ---------------------------------------------------------------------------
// Kernel 0: pack W_Q/W_K/W_V (f32 [1024,64] row-major) into per-lane B-fragment
// order, bf16: Wp[s][n][khalf][c] = W[(s*32 + khalf*16 + c)*64 + n]
// ---------------------------------------------------------------------------
__global__ __launch_bounds__(256) void packW_kernel(
    const float* __restrict__ Wq, const float* __restrict__ Wk,
    const float* __restrict__ Wv, bf16_t* __restrict__ out)
{
    const int PER = HIDDEN * HEAD;            // 65536
    int i = blockIdx.x * blockDim.x + threadIdx.x;
    if (i >= 3 * PER) return;
    int m = i / PER, j = i % PER;
    const float* W = (m == 0) ? Wq : ((m == 1) ? Wk : Wv);
    int c = j & 15, khalf = (j >> 4) & 1, n = (j >> 5) & 63, s = j >> 11;
    int k = s * 32 + khalf * 16 + c;
    out[(size_t)m * PER + j] = (bf16_t)W[k * HEAD + n];
}

// ---------------------------------------------------------------------------
// Kernel 1: Q,K,V projections. One wave per 16-row tile of x.
// Q,K stored bf16 row-major [B*T, 64]; V stored transposed bf16 [B][64][T].
// B-fragment loads are software-pipelined one tile ahead of the WMMAs.
// ---------------------------------------------------------------------------
__global__ __launch_bounds__(256) void proj_kernel(
    const float*  __restrict__ x,  const bf16_t* __restrict__ Wp,
    bf16_t* __restrict__ Qb, bf16_t* __restrict__ Kb, bf16_t* __restrict__ Vt)
{
    const int lane  = threadIdx.x & 31;
    const int wid   = (blockIdx.x * blockDim.x + threadIdx.x) >> 5;
    const int lrow  = lane & 15, lhalf = lane >> 4;
    const int rowbase = wid * 16;

    const bf16_t* WpQ = Wp;
    const bf16_t* WpK = Wp + HIDDEN * HEAD;
    const bf16_t* WpV = Wp + 2 * HIDDEN * HEAD;
    const float*  xrow = x + (size_t)(rowbase + lrow) * HIDDEN;

    v8f accQ[4], accK[4], accV[4];
    #pragma unroll
    for (int t = 0; t < 4; ++t)
        #pragma unroll
        for (int r = 0; r < 8; ++r) { accQ[t][r] = 0.f; accK[t][r] = 0.f; accV[t][r] = 0.f; }

    const int wbase = (lrow << 5) + (lhalf << 4);   // per-lane (loop-invariant) part of Wp offset

    for (int s = 0; s < HIDDEN / 32; ++s) {
        const int koff = s * 32 + lhalf * 16;
        v16bf a;
        #pragma unroll
        for (int c = 0; c < 8; ++c) a[c]     = (bf16_t)xrow[koff + c];
        #pragma unroll
        for (int c = 0; c < 8; ++c) a[8 + c] = (bf16_t)xrow[koff + 8 + c];

        // pipeline: issue tile t+1 loads before tile t WMMAs
        int off = (s << 11) + wbase;
        v16bf bq = loadB16(WpQ + off), bk = loadB16(WpK + off), bv = loadB16(WpV + off);
        #pragma unroll
        for (int t = 0; t < 4; ++t) {
            v16bf nbq, nbk, nbv;
            if (t < 3) {
                const int no = (s << 11) + ((t + 1) << 9) + wbase;   // (t+1)*16 rows * 32
                nbq = loadB16(WpQ + no);
                nbk = loadB16(WpK + no);
                nbv = loadB16(WpV + no);
            }
            accQ[t] = wmma_bf16(a, bq, accQ[t]);
            accK[t] = wmma_bf16(a, bk, accK[t]);
            accV[t] = wmma_bf16(a, bv, accV[t]);
            bq = nbq; bk = nbk; bv = nbv;
        }
    }

    const int b    = rowbase / TLEN;
    const int tpos = rowbase % TLEN;
    #pragma unroll
    for (int t = 0; t < 4; ++t)
        #pragma unroll
        for (int r = 0; r < 8; ++r) {
            const int M = r + lhalf * 8;            // C layout: row
            const int N = (t << 4) + lrow;          // C layout: col
            const size_t row = (size_t)rowbase + M;
            Qb[row * HEAD + N] = (bf16_t)accQ[t][r];
            Kb[row * HEAD + N] = (bf16_t)accK[t][r];
            Vt[((size_t)b * HEAD + N) * TLEN + tpos + M] = (bf16_t)accV[t][r];
        }
}

// ---------------------------------------------------------------------------
// Kernel 2: causal flash attention. One wave per 16-query tile, 32-key blocks.
// K (4KB) and V^T (4KB) blocks are double-buffered in LDS via async copies:
// next block's 16 async B128 copies are issued before computing on the
// current block, so VMEM latency overlaps softmax + WMMA work.
// Per-wave LDS slot: K[2][2048] + V[2][2048] + P[512] bf16 = 17408 B.
// ---------------------------------------------------------------------------
#define SLOT_ELEMS (2*2048 + 2*2048 + 512)

__global__ __launch_bounds__(256) void attn_kernel(
    const bf16_t* __restrict__ Qb, const bf16_t* __restrict__ Kb,
    const bf16_t* __restrict__ Vt, float* __restrict__ out)
{
    extern __shared__ char smem[];

    const int lane  = threadIdx.x & 31;
    const int wslot = threadIdx.x >> 5;
    const int wid   = blockIdx.x * 8 + wslot;
    const int lrow  = lane & 15, lhalf = lane >> 4;
    const int b     = wid / (TLEN / 16);
    const int qbase = (wid % (TLEN / 16)) * 16;
    const int koff8 = lhalf * 8;

    bf16_t* slot  = (bf16_t*)smem + (size_t)wslot * SLOT_ELEMS;
    bf16_t* kbufs = slot;                 // [2][32 rows * 64]
    bf16_t* vbufs = slot + 4096;          // [2][64 rows * 32]
    bf16_t* pl    = slot + 8192;          // [16 rows * 32]

    const bf16_t* Qrow  = Qb + ((size_t)b * TLEN + qbase + lrow) * HEAD;
    const v16bf aq0 = loadA16(Qrow, koff8);        // head dims 0..31
    const v16bf aq1 = loadA16(Qrow + 32, koff8);   // head dims 32..63
    const bf16_t* Kbase = Kb + (size_t)b * TLEN * HEAD;
    const bf16_t* Vbase = Vt + (size_t)b * HEAD * TLEN;

    // issue the 16 async B128 copies for key block kb into LDS buffer d
    auto issue_kv = [&](int kb, int d) {
        const char* gk = (const char*)(Kbase + (size_t)kb * HEAD);   // 4KB contiguous
        const unsigned lk = lds_off32(kbufs + d * 2048);
        #pragma unroll
        for (int i = 0; i < 8; ++i)
            async_copy_b128(lk + i * 512 + lane * 16, gk + i * 512 + lane * 16);
        const unsigned lv = lds_off32(vbufs + d * 2048);
        #pragma unroll
        for (int i = 0; i < 8; ++i) {
            const int j  = i * 32 + lane;          // 16B chunk id, 0..255
            const int dd = j >> 2, part = j & 3;   // head dim row, 16B quarter
            const char* gv = (const char*)(Vbase + (size_t)dd * TLEN + kb) + part * 16;
            async_copy_b128(lv + dd * 64 + part * 16, gv);
        }
    };

    float m[8], l[8];
    v8f acc[4];
    #pragma unroll
    for (int r = 0; r < 8; ++r) { m[r] = -1e30f; l[r] = 0.f; }
    #pragma unroll
    for (int t = 0; t < 4; ++t)
        #pragma unroll
        for (int r = 0; r < 8; ++r) acc[t][r] = 0.f;

    const int kend = qbase + 16;                   // exclusive key bound
    const int nblk = (kend + 31) / 32;

    issue_kv(0, 0);
    for (int ib = 0; ib < nblk; ++ib) {
        const int kb  = ib * 32;
        const int cur = ib & 1;

        wait_ds0();                                // prior LDS reads done (WAR safety)
        if (ib + 1 < nblk) {
            issue_kv(kb + 32, cur ^ 1);            // prefetch next block
            wait_async_le16();                     // current block's 16 copies complete
        } else {
            wait_async_0();
        }
        const bf16_t* kd = kbufs + cur * 2048;     // [32 keys][64 dims]
        const bf16_t* vd = vbufs + cur * 2048;     // [64 dims][32 keys]

        v8f s0, s1;
        #pragma unroll
        for (int r = 0; r < 8; ++r) { s0[r] = 0.f; s1[r] = 0.f; }

        // S = Q * K^T, two 16-key tiles per block (fragments from LDS)
        const bf16_t* k0 = kd + lrow * HEAD + lhalf * 16;
        s0 = wmma_bf16(aq0, loadB16(k0), s0);
        s0 = wmma_bf16(aq1, loadB16(k0 + 32), s0);
        const bool t1 = (kb + 16) < kend;
        if (t1) {
            const bf16_t* k1 = kd + (16 + lrow) * HEAD + lhalf * 16;
            s1 = wmma_bf16(aq0, loadB16(k1), s1);
            s1 = wmma_bf16(aq1, loadB16(k1 + 32), s1);
        }

        // online softmax (row = r + 8*lhalf; 16-lane reductions stay in-half)
        float corr[8];
        #pragma unroll
        for (int r = 0; r < 8; ++r) {
            const int q = qbase + r + lhalf * 8;
            float v0 = s0[r] * 0.125f;                         // 1/sqrt(64)
            if (kb + lrow > q) v0 = -1e30f;
            float v1 = t1 ? s1[r] * 0.125f : -1e30f;
            if (kb + 16 + lrow > q) v1 = -1e30f;

            float rmax = fmaxf(v0, v1);
            rmax = fmaxf(rmax, __shfl_xor(rmax, 1));
            rmax = fmaxf(rmax, __shfl_xor(rmax, 2));
            rmax = fmaxf(rmax, __shfl_xor(rmax, 4));
            rmax = fmaxf(rmax, __shfl_xor(rmax, 8));
            const float mnew = fmaxf(m[r], rmax);
            corr[r] = __expf(m[r] - mnew);
            const float p0 = __expf(v0 - mnew);
            const float p1 = __expf(v1 - mnew);
            float rs = p0 + p1;
            rs += __shfl_xor(rs, 1); rs += __shfl_xor(rs, 2);
            rs += __shfl_xor(rs, 4); rs += __shfl_xor(rs, 8);
            l[r] = l[r] * corr[r] + rs;
            m[r] = mnew;

            const int qrow = r + lhalf * 8;        // C-layout -> row-major P in LDS
            pl[qrow * 32 + lrow]      = (bf16_t)p0;
            pl[qrow * 32 + 16 + lrow] = (bf16_t)p1;
        }
        #pragma unroll
        for (int t = 0; t < 4; ++t)
            #pragma unroll
            for (int r = 0; r < 8; ++r) acc[t][r] *= corr[r];

        wait_ds0();                                // P writes visible to fragment read
        const v16bf pa = loadA16(pl + lrow * 32, koff8);

        // O += P * V  (V^T layout in LDS -> contiguous B fragments)
        #pragma unroll
        for (int t = 0; t < 4; ++t)
            acc[t] = wmma_bf16(pa, loadB16(vd + ((t << 4) + lrow) * 32 + lhalf * 16), acc[t]);
    }

    #pragma unroll
    for (int t = 0; t < 4; ++t)
        #pragma unroll
        for (int r = 0; r < 8; ++r) {
            const int q = qbase + r + lhalf * 8;
            out[((size_t)b * TLEN + q) * HEAD + (t << 4) + lrow] = acc[t][r] / l[r];
        }
}

// ---------------------------------------------------------------------------
extern "C" void kernel_launch(void* const* d_in, const int* in_sizes, int n_in,
                              void* d_out, int out_size, void* d_ws, size_t ws_size,
                              hipStream_t stream)
{
    const float* x  = (const float*)d_in[0];
    const float* Wq = (const float*)d_in[1];
    const float* Wk = (const float*)d_in[2];
    const float* Wv = (const float*)d_in[3];
    float* out = (float*)d_out;

    char* ws = (char*)d_ws;
    const size_t WP_BYTES = (size_t)3 * HIDDEN * HEAD * sizeof(bf16_t);  // 384 KB
    const size_t QKV_ELEM = (size_t)BATCH * TLEN * HEAD;                 // 1 Mi elems
    bf16_t* Wp = (bf16_t*)ws;
    bf16_t* Qb = (bf16_t*)(ws + WP_BYTES);
    bf16_t* Kb = (bf16_t*)(ws + WP_BYTES + QKV_ELEM * sizeof(bf16_t));
    bf16_t* Vt = (bf16_t*)(ws + WP_BYTES + 2 * QKV_ELEM * sizeof(bf16_t));

    // 1) pack W into WMMA B-fragment order (bf16)
    {
        int n = 3 * HIDDEN * HEAD;
        packW_kernel<<<(n + 255) / 256, 256, 0, stream>>>(Wq, Wk, Wv, Wp);
    }
    // 2) Q/K/V projections: 1024 waves (one per 16-token tile)
    {
        int waves = (BATCH * TLEN) / 16;           // 1024
        proj_kernel<<<(waves * 32) / 256, 256, 0, stream>>>(x, Wp, Qb, Kb, Vt);
    }
    // 3) causal flash attention: 1024 waves, 8 per block, async K/V staging
    {
        int waves = (BATCH * TLEN) / 16;           // 1024
        size_t shmem = (size_t)8 * SLOT_ELEMS * sizeof(bf16_t);  // 139264 B
        attn_kernel<<<(waves * 32) / 256, 256, shmem, stream>>>(Qb, Kb, Vt, out);
    }
}
